// OutputPPBlock_64080912056987
// MI455X (gfx1250) — compile-verified, matrix operand order checked by reading, not run on hardware
//
#include <hip/hip_runtime.h>
#include <hip/hip_bf16.h>

// ---------- types ----------
typedef __bf16 bf16_t;
typedef __attribute__((ext_vector_type(16))) __bf16 v16bf;
typedef __attribute__((ext_vector_type(8)))  __bf16 v8bf;
typedef __attribute__((ext_vector_type(4)))  __bf16 v4bf;
typedef __attribute__((ext_vector_type(8)))  float  v8f;
typedef __attribute__((ext_vector_type(4)))  unsigned int u32x4;

// Types matching the async-to-LDS builtin signature discovered from the
// round-2 diagnostic: param0 = (as1) int __vector_size__(16) *, non-const.
typedef int i32x4_v __attribute__((vector_size(16)));
typedef __attribute__((address_space(1))) i32x4_v* as1_i32x4p;
typedef __attribute__((address_space(3))) i32x4_v* as3_i32x4p;

// gfx1250 async direct-to-LDS path (probe via __has_builtin; fallback = VGPR copy)
#if defined(__HIP_DEVICE_COMPILE__) && defined(__gfx1250__) &&            \
    __has_builtin(__builtin_amdgcn_global_load_async_to_lds_b128) &&      \
    __has_builtin(__builtin_amdgcn_s_wait_asynccnt)
#define USE_ASYNC_LDS 1
#else
#define USE_ASYNC_LDS 0
#endif

// ---------- problem constants (match reference) ----------
constexpr int B_   = 4;
constexpr int E_   = 100000;
constexpr int N_   = 10000;
constexpr int NR_  = 6;
constexpr int H_   = 256;
constexpr int D_   = 256;
constexpr int OUT_ = 128;

constexpr int M_    = B_ * N_;   // 40000 MLP rows
constexpr int MT_   = 128;       // rows per workgroup (8 waves x 16)
constexpr int HPAD  = 264;       // 256 + 8 bf16 pad (16B) to dodge bank conflicts

// ws layout: [agg f32 B*N*H][wt bf16: Wup_t, Ws_t x3, Wout_t]
constexpr size_t AGG_ELEMS = (size_t)B_ * N_ * H_;          // 10,240,000 f32
constexpr size_t AGG_BYTES = AGG_ELEMS * sizeof(float);     // 40.96 MB (L2-resident)
constexpr int    WT_SQ     = 4 * 256 * 256;                 // Wup + 3x Ws  (262144)
constexpr int    WT_TOTAL  = WT_SQ + 128 * 256;             // + Wout_t     (294912)

// ---------------------------------------------------------------------------
// Kernel 0a: zero the scatter accumulator (must run every call).
// ---------------------------------------------------------------------------
__global__ void zero_f4_kernel(float4* __restrict__ p, int n4) {
    int t = blockIdx.x * blockDim.x + threadIdx.x;
    if (t < n4) p[t] = make_float4(0.f, 0.f, 0.f, 0.f);
}

// ---------------------------------------------------------------------------
// Kernel 0b: convert weights f32 -> bf16 and transpose to [out][in] so the
// WMMA B-fragment reads are contiguous along K.
// ---------------------------------------------------------------------------
__global__ void prep_weights_kernel(const float* __restrict__ Wup,
                                    const float* __restrict__ Ws,
                                    const float* __restrict__ Wout,
                                    bf16_t* __restrict__ wt) {
    int t = blockIdx.x * 256 + threadIdx.x;
    if (t < WT_SQ) {
        int l = t >> 16;            // 0..3
        int r = t & 65535;
        int n = r >> 8;             // out column
        int k = r & 255;            // in row
        float v = (l == 0) ? Wup[k * 256 + n] : Ws[(l - 1) * 65536 + k * 256 + n];
        wt[l * 65536 + n * 256 + k] = (bf16_t)v;
    } else if (t < WT_TOTAL) {
        int r = t - WT_SQ;          // 0..32767
        int n = r >> 8;             // 0..127
        int k = r & 255;
        wt[WT_SQ + n * 256 + k] = (bf16_t)Wout[k * 128 + n];
    }
}

// ---------------------------------------------------------------------------
// Kernel 1: g = (rbf @ W_rbf) * x, scatter-add over edges into agg[b,node,h].
// One 256-thread block per edge; 4 f32 atomics per lane (agg is L2-resident).
// HBM-bound on reading x (410 MB -> ~18 us floor at 23.3 TB/s).
// ---------------------------------------------------------------------------
__global__ __launch_bounds__(256) void scatter_kernel(
        const float* __restrict__ x, const float* __restrict__ rbf,
        const int* __restrict__ idx, const float* __restrict__ Wrbf,
        float* __restrict__ agg) {
    const int e = blockIdx.x;
    const int c = threadIdx.x;          // feature column 0..255

    // K=6 radial projection, per-edge uniform rbf values
    const float r0 = rbf[e * NR_ + 0], r1 = rbf[e * NR_ + 1], r2 = rbf[e * NR_ + 2];
    const float r3 = rbf[e * NR_ + 3], r4 = rbf[e * NR_ + 4], r5 = rbf[e * NR_ + 5];
    float s = r0 * Wrbf[0 * H_ + c] + r1 * Wrbf[1 * H_ + c] + r2 * Wrbf[2 * H_ + c]
            + r3 * Wrbf[3 * H_ + c] + r4 * Wrbf[4 * H_ + c] + r5 * Wrbf[5 * H_ + c];

    const int node = idx[e];

    if (e + 1 < E_)  // pull next edge's x line toward the WGP (global_prefetch_b8)
        __builtin_prefetch(&x[(size_t)(e + 1) * H_ + c], 0, 1);

#pragma unroll
    for (int b = 0; b < B_; ++b) {
        float g = s * x[((size_t)b * E_ + e) * (size_t)H_ + c];
        __hip_atomic_fetch_add(&agg[((size_t)b * N_ + node) * (size_t)H_ + c], g,
                               __ATOMIC_RELAXED, __HIP_MEMORY_SCOPE_AGENT);
    }
}

// ---------------------------------------------------------------------------
// Kernel 2: fused node-MLP via v_wmma_f32_16x16x32_bf16.
//   h = agg @ Wup ; 3x h = silu(h@W + b) ; out = h @ Wout
// ---------------------------------------------------------------------------

// Load a 16-element bf16 fragment per the CDNA5 16-bit operand layout:
// elements 0..7  = K kb..kb+7, elements 8..15 = K kb+16..kb+23 (kb = 0 or 8).
__device__ inline v16bf load_frag16(const bf16_t* p) {
    v8bf lo = *reinterpret_cast<const v8bf*>(p);        // 16B ds_load
    v8bf hi = *reinterpret_cast<const v8bf*>(p + 16);   // 16B ds_load
    return __builtin_shufflevector(lo, hi, 0, 1, 2, 3, 4, 5, 6, 7,
                                           8, 9, 10, 11, 12, 13, 14, 15);
}

// 16-byte global -> LDS copy: async direct-to-LDS when available.
__device__ inline void copy16_g2l(const bf16_t* __restrict__ g, bf16_t* l) {
#if USE_ASYNC_LDS
    __builtin_amdgcn_global_load_async_to_lds_b128(
        (as1_i32x4p)g, (as3_i32x4p)l, 0, 0);
#else
    *reinterpret_cast<u32x4*>(l) = *reinterpret_cast<const u32x4*>(g);
#endif
}

__device__ inline void wait_async_lds() {
#if USE_ASYNC_LDS
    __builtin_amdgcn_s_wait_asynccnt(0);
#endif
}

// Cooperative stage of one 16(out) x 256(K) bf16 weight tile (8 KB) into LDS.
__device__ inline void stage_w(const bf16_t* __restrict__ src,
                               bf16_t (&dst)[16][HPAD], int tid) {
    int rr = tid >> 4;              // 0..15 : out column within tile
    int cc = (tid & 15) * 16;       // 0..240: K offset, 32B per thread
    copy16_g2l(src + rr * 256 + cc,     &dst[rr][cc]);
    copy16_g2l(src + rr * 256 + cc + 8, &dst[rr][cc + 8]);
}

__global__ __launch_bounds__(256) void mlp_kernel(
        const float* __restrict__ agg, const bf16_t* __restrict__ wt,
        const float* __restrict__ bs, float* __restrict__ out) {
    __shared__ alignas(16) bf16_t hbuf[8][16][HPAD];   // 66 KB: per-wave activations
    __shared__ alignas(16) bf16_t wbuf[2][16][HPAD];   // 16.5 KB: weight tile dbl-buffer
    __shared__ float bias_s[256];

    const int tid  = threadIdx.x;
    const int wave = tid >> 5;
    const int lane = tid & 31;
    const int m16  = lane & 15;                 // row within 16 (A) / column N (B)
    const int kb   = (lane < 16) ? 0 : 8;       // K sub-offset per lane half
    const int mAdd = (lane >= 16) ? 8 : 0;      // C/D row offset per lane half
    const int rowBase = blockIdx.x * MT_ + wave * 16;

    // ---- load this wave's 16 agg rows (f32) -> hbuf (bf16), coalesced ----
    {
        const float4* a4 = reinterpret_cast<const float4*>(agg);
#pragma unroll
        for (int it = 0; it < 32; ++it) {
            int linear = it * 32 + lane;        // 16 rows x 64 float4
            int r = linear >> 6;
            int q = linear & 63;
            int gRow = rowBase + r;
            float4 v = (gRow < M_) ? a4[(size_t)gRow * 64 + q]
                                   : make_float4(0.f, 0.f, 0.f, 0.f);
            v4bf o;
            o[0] = (bf16_t)v.x; o[1] = (bf16_t)v.y;
            o[2] = (bf16_t)v.z; o[3] = (bf16_t)v.w;
            *reinterpret_cast<v4bf*>(&hbuf[wave][r][q * 4]) = o;
        }
    }

    // ---- 5-layer chain: l=0 Wup (linear), l=1..3 silu(hW+b), l=4 Wout ----
    for (int l = 0; l < 5; ++l) {
        // A fragments for this wave's 16 rows: 8 k-steps x 8 VGPRs
        v16bf afrag[8];
#pragma unroll
        for (int kf = 0; kf < 8; ++kf)
            afrag[kf] = load_frag16(&hbuf[wave][m16][kf * 32 + kb]);

        const bf16_t* wl = wt + ((l < 4) ? l * 65536 : WT_SQ);
        const int ntc = (l == 4) ? (OUT_ / 16) : (D_ / 16);
        const bool act = (l >= 1 && l <= 3);
        if (act) bias_s[tid] = bs[(l - 1) * 256 + tid];

        stage_w(wl, wbuf[0], tid);
        for (int nt = 0; nt < ntc; ++nt) {
            wait_async_lds();                   // async tile (nt) landed in LDS
            __syncthreads();                    // ... and is visible to all waves
            if (nt + 1 < ntc) stage_w(wl + (nt + 1) * 16 * 256, wbuf[(nt + 1) & 1], tid);

            const bf16_t (&wb)[16][HPAD] = wbuf[nt & 1];
            v8f c = {0.f, 0.f, 0.f, 0.f, 0.f, 0.f, 0.f, 0.f};
#pragma unroll
            for (int kf = 0; kf < 8; ++kf) {
                v16bf bfrag = load_frag16(&wb[m16][kf * 32 + kb]);
                c = __builtin_amdgcn_wmma_f32_16x16x32_bf16(
                        false, afrag[kf], false, bfrag, (short)0, c, false, false);
            }

            if (l < 4) {
                // C/D layout: VGPR r -> row (r + mAdd), lane%16 -> column
#pragma unroll
                for (int r = 0; r < 8; ++r) {
                    int m   = r + mAdd;
                    int col = nt * 16 + m16;
                    float v = c[r];
                    if (act) {
                        v += bias_s[col];
                        // silu = x * sigmoid(x); v_rcp_f32 instead of IEEE divide
                        v = v * __builtin_amdgcn_rcpf(1.f + __expf(-v));
                    }
                    hbuf[wave][m][col] = (bf16_t)v;
                }
            } else {
#pragma unroll
                for (int r = 0; r < 8; ++r) {
                    int m    = r + mAdd;
                    int gRow = rowBase + m;
                    if (gRow < M_)
                        out[(size_t)gRow * OUT_ + nt * 16 + m16] = c[r];
                }
            }
        }
        __syncthreads();   // layer epilogues done before bias_s / wbuf reuse
    }
}

// ---------------------------------------------------------------------------
// Host launcher
// ---------------------------------------------------------------------------
extern "C" void kernel_launch(void* const* d_in, const int* in_sizes, int n_in,
                              void* d_out, int out_size, void* d_ws, size_t ws_size,
                              hipStream_t stream) {
    const float* x    = (const float*)d_in[0];   // [B,E,H]
    const float* rbf  = (const float*)d_in[1];   // [E,NR]
    const int*   idx  = (const int*)  d_in[2];   // [E]
    const float* Wrbf = (const float*)d_in[3];   // [NR,H]
    const float* Wup  = (const float*)d_in[4];   // [H,D]
    const float* Ws   = (const float*)d_in[5];   // [L,D,D]
    const float* bs   = (const float*)d_in[6];   // [L,D]
    const float* Wout = (const float*)d_in[7];   // [D,OUT]
    float* out = (float*)d_out;

    float*  agg = (float*)d_ws;                                  // 40.96 MB
    bf16_t* wt  = (bf16_t*)((char*)d_ws + AGG_BYTES);            // 576 KB

    const int n4 = (int)(AGG_ELEMS / 4);
    zero_f4_kernel<<<(n4 + 255) / 256, 256, 0, stream>>>((float4*)agg, n4);
    prep_weights_kernel<<<(WT_TOTAL + 255) / 256, 256, 0, stream>>>(Wup, Ws, Wout, wt);
    scatter_kernel<<<E_, 256, 0, stream>>>(x, rbf, idx, Wrbf, agg);
    mlp_kernel<<<(M_ + MT_ - 1) / MT_, 256, 0, stream>>>(agg, wt, bs, out);
}
